// EncoderLSTM_A_35038343200805
// MI455X (gfx1250) — compile-verified, matrix operand order checked by reading, not run on hardware
//
#include <hip/hip_runtime.h>

#define B_ 128
#define T_ 1152
#define D_ 128
#define H_ 32
#define G_ 128   // 4*H_

typedef __attribute__((ext_vector_type(16))) __bf16 v16bf;
typedef __attribute__((ext_vector_type(8)))  float  v8f;
typedef __attribute__((ext_vector_type(4)))  unsigned int v4u;
typedef __attribute__((ext_vector_type(8)))  int    v8i;
typedef __attribute__((ext_vector_type(4)))  int    v4i;

#ifndef __has_builtin
#define __has_builtin(x) 0
#endif
#if __has_builtin(__builtin_amdgcn_tensor_load_to_lds)
#define USE_TDM 1
#else
#define USE_TDM 0
#endif

__device__ __forceinline__ float sigf(float x) { return 1.0f / (1.0f + __expf(-x)); }

// ---------------------------------------------------------------------------
// Phase 1: input projections for both directions (memory-bound bf16 WMMA GEMM)
//   xp[dir][t][b][g] = sum_d x[b][t][d] * W_ih[dir][g][d] + b_ih[g] + b_hh[g]
// ---------------------------------------------------------------------------
__global__ __launch_bounds__(256) void lstm_proj_wmma(
    const float* __restrict__ x,
    const float* __restrict__ Wf,  const float* __restrict__ Wb,
    const float* __restrict__ bif, const float* __restrict__ bhf,
    const float* __restrict__ bib, const float* __restrict__ bhb,
    float* __restrict__ xp)
{
  const int t     = blockIdx.x;
  const int btile = blockIdx.y;
  const int dir   = blockIdx.z;
  const int wave  = threadIdx.x >> 5;
  const int lane  = threadIdx.x & 31;
  const int n     = lane & 15;            // M row (A) / N col (B,C) within tile
  const int khalf = (lane >> 4) << 3;     // lanes 16-31 hold the +8 K half
  const int gbase = wave << 4;

  const float* W  = dir ? Wb : Wf;
  const float* bi = dir ? bib : bif;
  const float* bh = dir ? bhb : bhf;

  const float* xrow = x + ((size_t)(btile*16 + n)*T_ + t)*D_;  // A: M = lane&15
  const float* wrow = W + (size_t)(gbase + n)*D_;              // B: N = lane&15

  v8f acc = {};
  #pragma unroll
  for (int kk = 0; kk < D_; kk += 32) {
    v16bf a, b;
    const int ka = kk + khalf;
    #pragma unroll
    for (int i = 0; i < 8; ++i) {
      a[i]   = (__bf16)xrow[ka + i];      // K = ka..ka+7
      a[8+i] = (__bf16)xrow[ka + 16 + i]; // K = ka+16..ka+23
      b[i]   = (__bf16)wrow[ka + i];
      b[8+i] = (__bf16)wrow[ka + 16 + i];
    }
    acc = __builtin_amdgcn_wmma_f32_16x16x32_bf16(false, a, false, b,
                                                  (short)0, acc, false, false);
  }

  const float bias = bi[gbase + n] + bh[gbase + n];
  float* op = xp + (((size_t)dir*T_ + t)*B_ + btile*16)*G_ + gbase + n;
  const int mbase = (lane < 16) ? 0 : 8;  // C/D: VGPR r -> M=r (lo) / M=8+r (hi)
  #pragma unroll
  for (int r = 0; r < 8; ++r)
    op[(size_t)(mbase + r)*G_] = acc[r] + bias;
}

// ---------------------------------------------------------------------------
// TDM helper: 1D copy of one contiguous 8 KB xp tile (2048 fp32) into LDS.
// D# packing per CDNA5 ISA:
//   g0 = { count=1, lds_addr, global_addr[31:0], global_addr[56:32] | type=2 }
//   g1 = { data_size=4B, tensor_dim0=2048, tensor_dim1=1, tile_dim0=2048,
//          tile_dim1/2=0, tensor_dim0_stride=2048 }
// ---------------------------------------------------------------------------
#if USE_TDM
__device__ __forceinline__ void tdm_load_tile(const float* gptr, unsigned lds_off) {
  unsigned long long ga = (unsigned long long)(size_t)gptr;
  v4u g0;
  g0[0] = 1u;                                            // count=1, user D#
  g0[1] = lds_off;                                       // lds_addr (bytes)
  g0[2] = (unsigned)(ga & 0xffffffffu);                  // global_addr lo
  g0[3] = (unsigned)((ga >> 32) & 0x01ffffffu) | (2u << 30);  // hi | type=2
  v8i g1;
  g1[0] = 0x20000;                 // data_size=2 (4 bytes); no multicast/pad
  g1[1] = (int)(2048u << 16);      // tensor_dim0[15:0]=2048 -> bits [79:64]... [63:48] lo16
  g1[2] = (int)(1u << 16);         // tensor_dim0 hi16=0 | tensor_dim1 lo16=1
  g1[3] = (int)(2048u << 16);      // tensor_dim1 hi16=0 | tile_dim0=2048
  g1[4] = 0;                       // tile_dim1=0 (unused), tile_dim2=0
  g1[5] = 2048;                    // tensor_dim0_stride lo32
  g1[6] = 0;                       // stride hi16 | tensor_dim1_stride lo16
  g1[7] = 0;
  v4i z = {0, 0, 0, 0};
#if __clang_major__ >= 23
  v8i z8 = {0, 0, 0, 0, 0, 0, 0, 0};
  __builtin_amdgcn_tensor_load_to_lds(g0, g1, z, z, z8, 0);
#else
  __builtin_amdgcn_tensor_load_to_lds(g0, g1, z, z, 0);
#endif
}
#endif

// ---------------------------------------------------------------------------
// Phase 2: sequential LSTM scan (latency-bound).
// One workgroup = (dir, 16-row batch tile); wave w owns gate cols [16w,16w+16).
// Per step: ONE v_wmma_f32_16x16x32_bf16 per wave (K = H = 32).
// xp tiles are double-buffered in LDS via tensor_load_to_lds (TDM), issued one
// step ahead by wave 0 and fenced with s_wait_tensorcnt + workgroup barrier.
// ---------------------------------------------------------------------------
__global__ __launch_bounds__(256) void lstm_scan_wmma(
    const float* __restrict__ xp,
    const float* __restrict__ Whf, const float* __restrict__ Whb,
    float* __restrict__ out)
{
  __shared__ float  lds_xp[2][16 * G_];  // double-buffered xp tiles (2 x 8 KB)
  __shared__ float  g_lds[16][G_ + 4];   // pre-activation gates (padded)
  __shared__ __bf16 h_lds[16][H_ + 8];   // h, row stride 80B (b128-aligned)

  const int btile = blockIdx.x;
  const int dir   = blockIdx.y;
  const int tid   = threadIdx.x;
  const int wave  = tid >> 5;
  const int lane  = tid & 31;
  const int n     = lane & 15;
  const int khalf = (lane >> 4) << 3;
  const int gbase = wave << 4;
  const int mbase = (lane < 16) ? 0 : 8;

  const float* Wh = dir ? Whb : Whf;     // (G_, H_), row-major
  const float* xp_base = xp + ((size_t)dir*T_)*B_*G_ + (size_t)btile*16*G_;

  // Loop-invariant B fragment: B[k][n] = W_hh[gbase+n][k]
  v16bf bfrag;
  {
    const float* wr = Wh + (size_t)(gbase + n)*H_;
    #pragma unroll
    for (int i = 0; i < 8; ++i) {
      bfrag[i]   = (__bf16)wr[khalf + i];
      bfrag[8+i] = (__bf16)wr[khalf + 16 + i];
    }
  }

  for (int i = tid; i < 16*(H_ + 8); i += 256) ((__bf16*)h_lds)[i] = (__bf16)0.0f;

  // Persistent cell state: 2 of the 16x32 (m, h) cells per thread.
  float c0 = 0.0f, c1 = 0.0f;
  const int m0 = tid >> 5,  hc = tid & 31;   // cell tid
  const int m1 = m0 + 8;                     // cell tid+256 (same h column)

#if USE_TDM
  const unsigned lds_xp_off = (unsigned)(size_t)&lds_xp[0][0]; // low 32b = LDS offset
  if (wave == 0) {   // prologue: fetch tile for s = 0
    const int t0 = dir ? (T_ - 1) : 0;
    tdm_load_tile(xp_base + (size_t)t0*B_*G_, lds_xp_off);
  }
#endif
  __syncthreads();

  int buf = 0;
  for (int s = 0; s < T_; ++s) {
    const int tio = dir ? (T_ - 1 - s) : s;  // t index for xp read / out write

#if USE_TDM
    if (wave == 0) __builtin_amdgcn_s_wait_tensorcnt(0); // tile s has landed
#endif
    __syncthreads();   // barrier A: tile ready + h(s-1) visible

#if !USE_TDM
    const float* xpt = xp_base + (size_t)tio*B_*G_ + gbase + n;
#endif

    // A fragment: h from LDS, already in 16-bit A layout (row = M, col = K)
    v16bf a;
    #pragma unroll
    for (int i = 0; i < 8; ++i) {
      a[i]   = h_lds[n][khalf + i];
      a[8+i] = h_lds[n][khalf + 16 + i];
    }

    // C accumulator from the LDS-resident projection tile
    v8f acc;
    #pragma unroll
    for (int r = 0; r < 8; ++r)
#if USE_TDM
      acc[r] = lds_xp[buf][(mbase + r)*G_ + gbase + n];
#else
      acc[r] = xpt[(size_t)(mbase + r)*G_];
#endif

#if USE_TDM
    if (wave == 0 && s + 1 < T_) {   // prefetch tile s+1 into the other buffer
      const int tn = dir ? (T_ - 2 - s) : (s + 1);
      tdm_load_tile(xp_base + (size_t)tn*B_*G_,
                    lds_xp_off + (unsigned)((buf ^ 1) * 16 * G_ * 4));
    }
#endif

    acc = __builtin_amdgcn_wmma_f32_16x16x32_bf16(false, a, false, bfrag,
                                                  (short)0, acc, false, false);

    #pragma unroll
    for (int r = 0; r < 8; ++r) g_lds[mbase + r][gbase + n] = acc[r];
    __syncthreads();   // barrier B: gates complete

    // Elementwise cell update (fp32; v_exp transcendentals co-exec with WMMA)
    float i0 = g_lds[m0][hc],        f0 = g_lds[m0][H_ + hc];
    float z0 = g_lds[m0][2*H_ + hc], o0 = g_lds[m0][3*H_ + hc];
    c0 = sigf(f0)*c0 + sigf(i0)*tanhf(z0);
    float hh0 = sigf(o0)*tanhf(c0);

    float i1 = g_lds[m1][hc],        f1 = g_lds[m1][H_ + hc];
    float z1 = g_lds[m1][2*H_ + hc], o1 = g_lds[m1][3*H_ + hc];
    c1 = sigf(f1)*c1 + sigf(i1)*tanhf(z1);
    float hh1 = sigf(o1)*tanhf(c1);

    h_lds[m0][hc] = (__bf16)hh0;
    h_lds[m1][hc] = (__bf16)hh1;

    // out[b][t][dir*H + h], shape (B, T, 2H) fp32
    out[((size_t)(btile*16 + m0)*T_ + tio)*(2*H_) + dir*H_ + hc] = hh0;
    out[((size_t)(btile*16 + m1)*T_ + tio)*(2*H_) + dir*H_ + hc] = hh1;

    buf ^= 1;
  }
}

extern "C" void kernel_launch(void* const* d_in, const int* in_sizes, int n_in,
                              void* d_out, int out_size, void* d_ws, size_t ws_size,
                              hipStream_t stream) {
  const float* x   = (const float*)d_in[0];
  const float* Wif = (const float*)d_in[1];
  const float* Whf = (const float*)d_in[2];
  const float* bif = (const float*)d_in[3];
  const float* bhf = (const float*)d_in[4];
  const float* Wib = (const float*)d_in[5];
  const float* Whb = (const float*)d_in[6];
  const float* bib = (const float*)d_in[7];
  const float* bhb = (const float*)d_in[8];
  float* out = (float*)d_out;
  float* xp  = (float*)d_ws;   // 2*T*B*G fp32 = 151 MB scratch (fits in L2)

  dim3 g1(T_, B_/16, 2);       // (t, batch tile, dir)
  lstm_proj_wmma<<<g1, 256, 0, stream>>>(x, Wif, Wib, bif, bhf, bib, bhb, xp);

  dim3 g2(B_/16, 2);           // (batch tile, dir)
  lstm_scan_wmma<<<g2, 256, 0, stream>>>(xp, Whf, Whb, out);

  (void)in_sizes; (void)n_in; (void)out_size; (void)ws_size;
}
